// RealNVP_62749472195096
// MI455X (gfx1250) — matrix operand order repaired and checked by reading, not compiled, not run on hardware
//
#include <hip/hip_runtime.h>
#include <hip/hip_bf16.h>
#include <math.h>

typedef __attribute__((ext_vector_type(16))) _Float16 v16h;
typedef __attribute__((ext_vector_type(8)))  float    v8f;

// Per-lane resident constants (weights are tiny: 18 KB total, loop-invariant).
struct MlpConsts {
    v16h  w1[4];   // layer1 weights, A-operand layout (f16)
    v16h  b1[4];   // layer1 bias,    A-operand layout (f16)
    v16h  B0[4];   // W2 B-operand, N-tile 0 (cols 0..15)
    v16h  B1[4];   // W2 B-operand, N-tile 1 (cols 16..31)
    float b2a[4], b2b[4];   // layer2 bias at col m and m+16
    float w3a[4], w3b[4];   // layer3 weight at row m and m+16
    float b3v[4];           // layer3 bias
};

// One MLP (1 -> 32 -> 32 -> 1) over 16 samples (rows). Input xm is the
// per-row scalar (each lane holds the value for row = lane%16). Returns the
// per-row scalar output, broadcast to every lane of that row.
template <int I>
__device__ __forceinline__ float mlp_eval(const MlpConsts& C, float xm,
                                          int lane, volatile float* vb) {
    // ---- layer 1: build A operand (16x32 f16) directly in WMMA A layout ----
    _Float16 xh = (_Float16)xm;
    v16h a;
#pragma unroll
    for (int t = 0; t < 16; ++t) {
        _Float16 h = xh * C.w1[I][t] + C.b1[I][t];
        a[t] = (h > (_Float16)0.0f) ? h : (_Float16)0.0f;
    }

    // ---- layer 2: (16x32) @ (32x32) = two 16x16x32 f16 WMMAs, f32 accum ----
    v8f cz = {};
    v8f c0 = __builtin_amdgcn_wmma_f32_16x16x32_f16(
        false, a, false, C.B0[I], (short)0, cz, false, false);
    v8f c1 = __builtin_amdgcn_wmma_f32_16x16x32_f16(
        false, a, false, C.B1[I], (short)0, cz, false, false);

    // ---- layer 3: bias+ReLU then weighted column reduction ----
    // C layout: reg r holds [row = r + 8*(lane>=16)][col = lane%16 (+16)]
    float partial[8];
#pragma unroll
    for (int r = 0; r < 8; ++r) {
        float ha = c0[r] + C.b2a[I]; ha = ha > 0.0f ? ha : 0.0f;
        float hb = c1[r] + C.b2b[I]; hb = hb > 0.0f ? hb : 0.0f;
        partial[r] = ha * C.w3a[I] + hb * C.w3b[I];
    }
    // butterfly over the 16 columns within each half-wave
#pragma unroll
    for (int mask = 1; mask <= 8; mask <<= 1) {
#pragma unroll
        for (int r = 0; r < 8; ++r)
            partial[r] += __shfl_xor(partial[r], mask, 32);
    }

    // ---- broadcast per-row outputs to all lanes via per-wave LDS slot ----
    int rbase = (lane < 16) ? 0 : 8;   // which rows this half-wave owns
#pragma unroll
    for (int r = 0; r < 8; ++r)
        vb[rbase + r] = partial[r] + C.b3v[I];
    __builtin_amdgcn_wave_barrier();
    float out = vb[lane & 15];          // row m = lane%16
    __builtin_amdgcn_wave_barrier();
    return out;
}

__global__ void __launch_bounds__(256)
realnvp_kernel(const float* __restrict__ z,
               const float* __restrict__ W1, const float* __restrict__ b1,
               const float* __restrict__ W2, const float* __restrict__ b2,
               const float* __restrict__ W3, const float* __restrict__ b3,
               float* __restrict__ out, int Bn) {
    __shared__ float sbuf[8 * 16];                 // 16 floats per wave
    const int lane = threadIdx.x & 31;
    const int wib  = threadIdx.x >> 5;
    volatile float* vb = sbuf + wib * 16;

    const int m  = lane & 15;                      // row / column index 0..15
    const int hi = lane >> 4;                      // half-wave id

    // ---------------- preload loop-invariant weights into registers --------
    MlpConsts C;
    {
        const int kb8  = hi * 8;                   // A-layout K base
        const int kb16 = hi * 16;                  // unused; B reuses the A-layout K map
#pragma unroll
        for (int i = 0; i < 4; ++i) {
#pragma unroll
            for (int t = 0; t < 16; ++t) {
                int k = kb8 + (t < 8 ? t : t + 8);         // A layout K map
                C.w1[i][t] = (_Float16)W1[i * 32 + k];
                C.b1[i][t] = (_Float16)b1[i * 32 + k];
                int kk = k;                                // B layout K map (same as A)
                C.B0[i][t] = (_Float16)W2[i * 1024 + kk * 32 + m];
                C.B1[i][t] = (_Float16)W2[i * 1024 + kk * 32 + 16 + m];
            }
            C.b2a[i] = b2[i * 32 + m];
            C.b2b[i] = b2[i * 32 + 16 + m];
            C.w3a[i] = W3[i * 32 + m];
            C.w3b[i] = W3[i * 32 + 16 + m];
            C.b3v[i] = b3[i];
        }
    }

    const int ntiles = Bn >> 4;                    // 16 samples per tile
    const int gwave  = blockIdx.x * (blockDim.x >> 5) + wib;
    const int nwaves = gridDim.x * (blockDim.x >> 5);
    float* zout  = out;
    float* ldout = out + (size_t)Bn * 2;

    for (int tile = gwave; tile < ntiles; tile += nwaves) {
        const size_t base = (size_t)tile * 32;
        // 16 samples = 32 contiguous floats: fully coalesced single b32 load
        float zval = z[base + lane];
        // redistribute: every lane gets (z1, z2) of its row m
        float z1m = __shfl(zval, (m << 1) + 0, 32);
        float z2m = __shfl(zval, (m << 1) + 1, 32);

        // coupling layer 1: z2' = z2 * exp(MLP0(z1)) + MLP1(z1)
        float ldt1 = mlp_eval<0>(C, z1m, lane, vb);
        float t1   = mlp_eval<1>(C, z1m, lane, vb);
        float z2n  = z2m * expf(ldt1) + t1;

        // coupling layer 2: z1' = z1 * exp(MLP2(z2')) + MLP3(z2')
        float ldt2 = mlp_eval<2>(C, z2n, lane, vb);
        float t2   = mlp_eval<3>(C, z2n, lane, vb);
        float z1n  = z1m * expf(ldt2) + t2;
        float ldts = ldt1 + ldt2;

        // coalesced interleaved store of (z1', z2')
        int sm = lane >> 1;
        float o1 = __shfl(z1n, sm, 32);
        float o2 = __shfl(z2n, sm, 32);
        zout[base + lane] = (lane & 1) ? o2 : o1;
        // log_det: 16 contiguous floats per tile
        if (lane < 16)
            ldout[(size_t)tile * 16 + lane] = ldts;
    }
}

extern "C" void kernel_launch(void* const* d_in, const int* in_sizes, int n_in,
                              void* d_out, int out_size, void* d_ws, size_t ws_size,
                              hipStream_t stream) {
    const float* z  = (const float*)d_in[0];
    const float* W1 = (const float*)d_in[1];
    const float* b1 = (const float*)d_in[2];
    const float* W2 = (const float*)d_in[3];
    const float* b2 = (const float*)d_in[4];
    const float* W3 = (const float*)d_in[5];
    const float* b3 = (const float*)d_in[6];
    float* out = (float*)d_out;

    const int Bn = in_sizes[0] / 2;                // (B, 2) input
    const int threads = 256;                       // 8 waves / block
    int blocks = 2048;                             // grid-stride, 8 tiles/wave
    const int ntiles = Bn >> 4;
    const int maxBlocks = (ntiles + 7) / 8;        // don't overshoot tiny B
    if (blocks > maxBlocks && maxBlocks > 0) blocks = maxBlocks;

    realnvp_kernel<<<blocks, threads, 0, stream>>>(z, W1, b1, W2, b2, W3, b3,
                                                   out, Bn);
}
